// LTXSelfAttention_3375844295180
// MI455X (gfx1250) — matrix operand order repaired and checked by reading, not compiled, not run on hardware
//
#include <hip/hip_runtime.h>

// ---------------- problem constants ----------------
#define BB 2
#define SS 2048
#define DD 2048
#define HH 32
#define DH 64
#define EPSF 1e-6f

// GEMM blocking
#define GBLK_M 64
#define GBLK_N 256
#define GBLK_K 32

typedef __bf16 bf16;
typedef bf16 bf16x8  __attribute__((ext_vector_type(8)));
typedef bf16 bf16x16 __attribute__((ext_vector_type(16)));
typedef float floatx4 __attribute__((ext_vector_type(4)));
typedef float floatx8 __attribute__((ext_vector_type(8)));

// ---------------- WMMA helpers (CDNA5 layouts) ----------------
// A-matrix 16x32 bf16, row-major src (ld elems/row):
// lane (hi = lane>=16, r = lane&15): halves 0..7  = A[r, hi*8 + j]
//                                     halves 8..15 = A[r, 16 + hi*8 + j]
__device__ __forceinline__ bf16x16 load_frag_a(const bf16* base, int ld, int lane) {
    int hi = (lane >> 4) & 1;
    int r  = lane & 15;
    const bf16* p = base + (size_t)r * ld + hi * 8;
    bf16x8 lo = *(const bf16x8*)p;
    bf16x8 hh = *(const bf16x8*)(p + 16);
    return __builtin_shufflevector(lo, hh, 0,1,2,3,4,5,6,7,8,9,10,11,12,13,14,15);
}

// B-matrix 32x16 bf16 where B[k,n] = base[n*ld + k] (column n contiguous over k):
// lane (hi, n = lane&15): halves 0..7 = B[hi*8 + j, n], halves 8..15 = B[16 + hi*8 + j, n]
__device__ __forceinline__ bf16x16 load_frag_b(const bf16* base, int ld, int lane) {
    int hi = (lane >> 4) & 1;
    int n  = lane & 15;
    const bf16* p = base + (size_t)n * ld + hi * 8;
    bf16x8 lo = *(const bf16x8*)p;
    bf16x8 hh = *(const bf16x8*)(p + 16);
    return __builtin_shufflevector(lo, hh, 0,1,2,3,4,5,6,7,8,9,10,11,12,13,14,15);
}

__device__ __forceinline__ floatx8 wmma_bf16(bf16x16 a, bf16x16 b, floatx8 c) {
    return __builtin_amdgcn_wmma_f32_16x16x32_bf16(false, a, false, b, (short)0, c, false, false);
}

// ---------------- CDNA5 async copy: global -> LDS (ASYNCcnt-tracked) ----------------
__device__ __forceinline__ void async_load_b128(unsigned lds_off, const bf16* gaddr) {
    asm volatile("global_load_async_to_lds_b128 %0, %1, off"
                 :: "v"(lds_off), "v"(gaddr) : "memory");
}
__device__ __forceinline__ void wait_asynccnt0() {
    asm volatile("s_wait_asynccnt 0x0" ::: "memory");
}
__device__ __forceinline__ unsigned lds_offset(const void* p) {
    // shared aperture lives in the high 32 bits; low 32 bits are the LDS byte offset
    return (unsigned)(unsigned long long)p;
}

// ---------------- fp32 -> bf16 convert (vectorized x8) ----------------
__global__ void cvt_f32_bf16_kernel(const float* __restrict__ src, bf16* __restrict__ dst, int n8) {
    int i = blockIdx.x * blockDim.x + threadIdx.x;
    int stride = gridDim.x * blockDim.x;
    for (; i < n8; i += stride) {
        floatx4 a = *(const floatx4*)(src + (size_t)i * 8);
        floatx4 b = *(const floatx4*)(src + (size_t)i * 8 + 4);
        bf16x8 o;
#pragma unroll
        for (int j = 0; j < 4; ++j) { o[j] = (bf16)a[j]; o[4 + j] = (bf16)b[j]; }
        *(bf16x8*)(dst + (size_t)i * 8) = o;
    }
}

// ---------------- GEMM: C[m,n] = sum_k A[m,k]*W[n,k] + bias[n] ----------------
// A: MxK bf16 row-major; W: NxK bf16 row-major (B columns = W rows); C fp32.
// Block = 256 threads = 8 waves; block tile 64x256 staged in LDS via async copies
// (double-buffered); wave tile 32x64 (2 A-frags x 4 B-frags -> 8 WMMAs / K-step).
__global__ void gemm_bias_kernel(const bf16* __restrict__ A, const bf16* __restrict__ W,
                                 const float* __restrict__ bias, float* __restrict__ C,
                                 int M, int N, int K) {
    __shared__ __align__(16) bf16 As[2][GBLK_M][GBLK_K];   // 2 x 4 KB
    __shared__ __align__(16) bf16 Bs[2][GBLK_N][GBLK_K];   // 2 x 16 KB

    int tid  = threadIdx.x;
    int lane = tid & 31;
    int wave = tid >> 5;
    int mt = wave & 1;          // 2 M sub-tiles of 32
    int nt = wave >> 1;         // 4 N sub-tiles of 64
    int mblk = blockIdx.x * GBLK_M;
    int nblk = blockIdx.y * GBLK_N;

    // staging assignment: each thread moves one 16B chunk of A and four of B per K-step
    int srow = tid >> 2;        // 0..63
    int sseg = tid & 3;         // 0..3  (8 bf16 per segment)
    const bf16* agsrc = A + (size_t)(mblk + srow) * K + sseg * 8;
    const bf16* bgsrc = W + (size_t)(nblk + srow) * K + sseg * 8;

    auto stage = [&](int buf, int k0) {
        async_load_b128(lds_offset(&As[buf][srow][sseg * 8]), agsrc + k0);
#pragma unroll
        for (int i = 0; i < 4; ++i) {
            async_load_b128(lds_offset(&Bs[buf][srow + i * 64][sseg * 8]),
                            bgsrc + (size_t)(i * 64) * K + k0);
        }
    };

    floatx8 acc[2][4] = {};

    stage(0, 0);
    wait_asynccnt0();
    __syncthreads();

    int m0w = mt * 32;
    int n0w = nt * 64;
    for (int k0 = 0; k0 < K; k0 += GBLK_K) {
        int cur = (k0 >> 5) & 1;
        if (k0 + GBLK_K < K) stage(cur ^ 1, k0 + GBLK_K);

        bf16x16 a0 = load_frag_a(&As[cur][m0w][0],      GBLK_K, lane);
        bf16x16 a1 = load_frag_a(&As[cur][m0w + 16][0], GBLK_K, lane);
#pragma unroll
        for (int f = 0; f < 4; ++f) {
            bf16x16 b = load_frag_b(&Bs[cur][n0w + f * 16][0], GBLK_K, lane);
            acc[0][f] = wmma_bf16(a0, b, acc[0][f]);
            acc[1][f] = wmma_bf16(a1, b, acc[1][f]);
        }
        wait_asynccnt0();
        __syncthreads();
    }

    int hi = (lane >> 4) & 1;
    int n  = lane & 15;
#pragma unroll
    for (int mi = 0; mi < 2; ++mi) {
#pragma unroll
        for (int f = 0; f < 4; ++f) {
            int col = nblk + n0w + f * 16 + n;
            float bvl = bias[col];
#pragma unroll
            for (int r = 0; r < 8; ++r) {
                int row = mblk + m0w + mi * 16 + r + hi * 8;
                C[(size_t)row * N + col] = acc[mi][f][r] + bvl;
            }
        }
    }
}

// ---------------- RMSNorm + interleaved RoPE + head repack ----------------
// qf/kf/vf: [B*S, D] fp32. Outputs: qh,kh: [B,H,S,DH] bf16; vhT: [B,H,DH,S] bf16.
__global__ void norm_rope_reorg_kernel(const float* __restrict__ qf, const float* __restrict__ kf,
                                       const float* __restrict__ vf,
                                       const float* __restrict__ qw, const float* __restrict__ kw,
                                       const float* __restrict__ cosp, const float* __restrict__ sinp,
                                       bf16* __restrict__ qh, bf16* __restrict__ kh,
                                       bf16* __restrict__ vhT) {
    int row = blockIdx.x;             // b*S + s
    int b = row / SS;
    int s = row % SS;
    int tid = threadIdx.x;            // 256 threads, 8 elems each
    int d0 = tid * 8;
    int lane = tid & 31;
    int wave = tid >> 5;

    const float* qr = qf + (size_t)row * DD;
    const float* kr = kf + (size_t)row * DD;
    const float* vr = vf + (size_t)row * DD;

    float qv[8], kv[8], vv[8];
    float ssq_q = 0.f, ssq_k = 0.f;
#pragma unroll
    for (int i = 0; i < 8; ++i) {
        qv[i] = qr[d0 + i];
        kv[i] = kr[d0 + i];
        vv[i] = vr[d0 + i];
        ssq_q += qv[i] * qv[i];
        ssq_k += kv[i] * kv[i];
    }
#pragma unroll
    for (int off = 16; off > 0; off >>= 1) {
        ssq_q += __shfl_xor(ssq_q, off, 32);
        ssq_k += __shfl_xor(ssq_k, off, 32);
    }
    __shared__ float redq[8];
    __shared__ float redk[8];
    if (lane == 0) { redq[wave] = ssq_q; redk[wave] = ssq_k; }
    __syncthreads();
    float tq = 0.f, tk = 0.f;
#pragma unroll
    for (int i = 0; i < 8; ++i) { tq += redq[i]; tk += redk[i]; }
    float rq = rsqrtf(tq * (1.0f / DD) + EPSF);
    float rk = rsqrtf(tk * (1.0f / DD) + EPSF);

    const float* cs = cosp + (size_t)s * DD;
    const float* sn = sinp + (size_t)s * DD;

    int h   = d0 >> 6;
    int dh0 = d0 & 63;
    bf16* qdst = qh + (((size_t)(b * HH + h) * SS + s) * DH + dh0);
    bf16* kdst = kh + (((size_t)(b * HH + h) * SS + s) * DH + dh0);
    bf16* vdst = vhT + ((size_t)(b * HH + h) * DH + dh0) * SS + s;

#pragma unroll
    for (int i = 0; i < 8; i += 2) {
        int d = d0 + i;
        float c0 = cs[d], c1 = cs[d + 1];
        float s0 = sn[d], s1 = sn[d + 1];

        float y0 = qv[i] * rq * qw[d];
        float y1 = qv[i + 1] * rq * qw[d + 1];
        qdst[i]     = (bf16)(y0 * c0 - y1 * s0);
        qdst[i + 1] = (bf16)(y1 * c1 + y0 * s1);

        float z0 = kv[i] * rk * kw[d];
        float z1 = kv[i + 1] * rk * kw[d + 1];
        kdst[i]     = (bf16)(z0 * c0 - z1 * s0);
        kdst[i + 1] = (bf16)(z1 * c1 + z0 * s1);
    }
#pragma unroll
    for (int i = 0; i < 8; ++i) vdst[(size_t)i * SS] = (bf16)vv[i];
}

// ---------------- Flash attention ----------------
// qh,kh: [B,H,S,DH] bf16; vhT: [B,H,DH,S] bf16; ao: [B,S,D] bf16.
// One wave: 16 queries x full DH, online softmax over S keys; P transposed via LDS.
__global__ void flash_attn_kernel(const bf16* __restrict__ qh, const bf16* __restrict__ kh,
                                  const bf16* __restrict__ vhT, bf16* __restrict__ ao) {
    __shared__ __align__(16) bf16 ptile[8][16 * 32];

    int lane = threadIdx.x & 31;
    int wave = threadIdx.x >> 5;
    int gw = blockIdx.x * 8 + wave;
    int qt = gw & 127;                     // S/16 = 128 query tiles
    int bh = gw >> 7;                      // 0..B*H-1
    int q0 = qt * 16;
    int h = bh & (HH - 1);
    int b = bh >> 5;

    const bf16* qbase = qh + ((size_t)bh * SS + q0) * DH;
    const bf16* kbase = kh + (size_t)bh * SS * DH;
    const bf16* vbase = vhT + (size_t)bh * DH * SS;

    bf16x16 aq0 = load_frag_a(qbase + 0,  DH, lane);
    bf16x16 aq1 = load_frag_a(qbase + 32, DH, lane);

    floatx8 o0 = {}, o1 = {}, o2 = {}, o3 = {};
    float mrow[8], lrow[8];
#pragma unroll
    for (int r = 0; r < 8; ++r) { mrow[r] = -3.0e38f; lrow[r] = 0.f; }

    int hi = (lane >> 4) & 1;
    int nn = lane & 15;
    bf16* pw = &ptile[wave][0];
    const float scale = 0.125f;  // 1/sqrt(64)

    for (int kb = 0; kb < SS; kb += 32) {
        floatx8 sc0 = {}, sc1 = {};
        {
            bf16x16 bk0a = load_frag_b(kbase + (size_t)kb * DH + 0,  DH, lane);
            bf16x16 bk0b = load_frag_b(kbase + (size_t)kb * DH + 32, DH, lane);
            sc0 = wmma_bf16(aq0, bk0a, sc0);
            sc0 = wmma_bf16(aq1, bk0b, sc0);
            bf16x16 bk1a = load_frag_b(kbase + (size_t)(kb + 16) * DH + 0,  DH, lane);
            bf16x16 bk1b = load_frag_b(kbase + (size_t)(kb + 16) * DH + 32, DH, lane);
            sc1 = wmma_bf16(aq0, bk1a, sc1);
            sc1 = wmma_bf16(aq1, bk1b, sc1);
        }
#pragma unroll
        for (int r = 0; r < 8; ++r) {
            float a = sc0[r] * scale;
            float c = sc1[r] * scale;
            float mx = fmaxf(a, c);
#pragma unroll
            for (int off = 1; off < 16; off <<= 1) mx = fmaxf(mx, __shfl_xor(mx, off, 32));
            float mnew = fmaxf(mrow[r], mx);
            float alpha = __expf(mrow[r] - mnew);
            float p0 = __expf(a - mnew);
            float p1 = __expf(c - mnew);
            float rs = p0 + p1;
#pragma unroll
            for (int off = 1; off < 16; off <<= 1) rs += __shfl_xor(rs, off, 32);
            lrow[r] = lrow[r] * alpha + rs;
            mrow[r] = mnew;
            o0[r] *= alpha; o1[r] *= alpha; o2[r] *= alpha; o3[r] *= alpha;
            int prow = r + hi * 8;
            pw[prow * 32 + nn]      = (bf16)p0;
            pw[prow * 32 + 16 + nn] = (bf16)p1;
        }
        asm volatile("s_wait_dscnt 0x0" ::: "memory");

        bf16x16 ap;
        {
            const bf16* pa = pw + (size_t)(lane & 15) * 32 + hi * 8;
            bf16x8 lo = *(const bf16x8*)pa;
            bf16x8 hh = *(const bf16x8*)(pa + 16);
            ap = __builtin_shufflevector(lo, hh, 0,1,2,3,4,5,6,7,8,9,10,11,12,13,14,15);
        }
        bf16x16 bv0 = load_frag_b(vbase + (size_t)0  * SS + kb, SS, lane);
        bf16x16 bv1 = load_frag_b(vbase + (size_t)16 * SS + kb, SS, lane);
        bf16x16 bv2 = load_frag_b(vbase + (size_t)32 * SS + kb, SS, lane);
        bf16x16 bv3 = load_frag_b(vbase + (size_t)48 * SS + kb, SS, lane);
        o0 = wmma_bf16(ap, bv0, o0);
        o1 = wmma_bf16(ap, bv1, o1);
        o2 = wmma_bf16(ap, bv2, o2);
        o3 = wmma_bf16(ap, bv3, o3);
    }

    bf16* obase = ao + ((size_t)b * SS + q0) * DD + h * 64;
    floatx8 os[4] = {o0, o1, o2, o3};
#pragma unroll
    for (int f = 0; f < 4; ++f) {
#pragma unroll
        for (int r = 0; r < 8; ++r) {
            float val = os[f][r] / lrow[r];
            obase[(size_t)(r + hi * 8) * DD + f * 16 + nn] = (bf16)val;
        }
    }
}

// ---------------- host-side orchestration ----------------
extern "C" void kernel_launch(void* const* d_in, const int* in_sizes, int n_in,
                              void* d_out, int out_size, void* d_ws, size_t ws_size,
                              hipStream_t stream) {
    const float* x      = (const float*)d_in[0];
    const float* pe_cos = (const float*)d_in[1];
    const float* pe_sin = (const float*)d_in[2];
    const float* Wq     = (const float*)d_in[3];
    const float* bq     = (const float*)d_in[4];
    const float* Wk     = (const float*)d_in[5];
    const float* bk     = (const float*)d_in[6];
    const float* Wv     = (const float*)d_in[7];
    const float* bv     = (const float*)d_in[8];
    const float* qnw    = (const float*)d_in[9];
    const float* knw    = (const float*)d_in[10];
    const float* Wo     = (const float*)d_in[11];
    const float* bo     = (const float*)d_in[12];

    const size_t SZ_X = (size_t)BB * SS * DD;  // 8M elems
    const size_t SZ_W = (size_t)DD * DD;       // 4M elems

    char* ws = (char*)d_ws;
    bf16* xb  = (bf16*)ws;                 ws += SZ_X * sizeof(bf16);
    bf16* Wqb = (bf16*)ws;                 ws += SZ_W * sizeof(bf16);
    bf16* Wkb = (bf16*)ws;                 ws += SZ_W * sizeof(bf16);
    bf16* Wvb = (bf16*)ws;                 ws += SZ_W * sizeof(bf16);
    bf16* Wob = (bf16*)ws;                 ws += SZ_W * sizeof(bf16);
    float* qf = (float*)ws;                ws += SZ_X * sizeof(float);
    float* kf = (float*)ws;                ws += SZ_X * sizeof(float);
    float* vf = (float*)ws;                ws += SZ_X * sizeof(float);
    bf16* qh  = (bf16*)ws;                 ws += SZ_X * sizeof(bf16);
    bf16* kh  = (bf16*)ws;                 ws += SZ_X * sizeof(bf16);
    bf16* vhT = (bf16*)ws;                 ws += SZ_X * sizeof(bf16);
    bf16* ao  = (bf16*)qf;                 // reuse qf region after stage 3

    // 1) convert inputs to bf16 (x8 vectorized)
    cvt_f32_bf16_kernel<<<1024, 256, 0, stream>>>(x,  xb,  (int)(SZ_X / 8));
    cvt_f32_bf16_kernel<<<512,  256, 0, stream>>>(Wq, Wqb, (int)(SZ_W / 8));
    cvt_f32_bf16_kernel<<<512,  256, 0, stream>>>(Wk, Wkb, (int)(SZ_W / 8));
    cvt_f32_bf16_kernel<<<512,  256, 0, stream>>>(Wv, Wvb, (int)(SZ_W / 8));
    cvt_f32_bf16_kernel<<<512,  256, 0, stream>>>(Wo, Wob, (int)(SZ_W / 8));

    // 2) QKV projections: [4096,2048] = xb * W^T + b
    dim3 ggrid(BB * SS / GBLK_M, DD / GBLK_N);  // 64 x 8
    gemm_bias_kernel<<<ggrid, 256, 0, stream>>>(xb, Wqb, bq, qf, BB * SS, DD, DD);
    gemm_bias_kernel<<<ggrid, 256, 0, stream>>>(xb, Wkb, bk, kf, BB * SS, DD, DD);
    gemm_bias_kernel<<<ggrid, 256, 0, stream>>>(xb, Wvb, bv, vf, BB * SS, DD, DD);

    // 3) RMSNorm + RoPE + head repack
    norm_rope_reorg_kernel<<<BB * SS, 256, 0, stream>>>(qf, kf, vf, qnw, knw,
                                                        pe_cos, pe_sin, qh, kh, vhT);

    // 4) flash attention: 8192 waves, 8 waves per block
    flash_attn_kernel<<<(BB * HH * (SS / 16)) / 8, 256, 0, stream>>>(qh, kh, vhT, ao);

    // 5) output projection -> fp32 d_out
    gemm_bias_kernel<<<ggrid, 256, 0, stream>>>(ao, Wob, bo, (float*)d_out, BB * SS, DD, DD);
}
